// SlidingWindowAttention_48120813584716
// MI455X (gfx1250) — matrix operand order repaired and checked by reading, not compiled, not run on hardware
//
#include <hip/hip_runtime.h>

typedef __bf16 bf16;
typedef __attribute__((ext_vector_type(16))) __bf16 v16bf;
typedef __attribute__((ext_vector_type(8)))  float  v8f;
typedef __attribute__((ext_vector_type(8)))  unsigned int v8u;
typedef unsigned int u32x4 __attribute__((ext_vector_type(4)));
typedef int          i32x8 __attribute__((ext_vector_type(8)));
typedef int          i32x4 __attribute__((ext_vector_type(4)));

#define N_HEADS  16
#define HEAD_DIM 64
#define N_EMBD   1024
#define SEQ_T    2048
#define BATCH    2
#define WINDOW   512
#define M_TOTAL  (BATCH * SEQ_T)   // 4096

#if __has_builtin(__builtin_amdgcn_tensor_load_to_lds)
#define HAVE_TDM 1
#endif

static __device__ __forceinline__ v8f vzero8() {
  v8f z;
#pragma unroll
  for (int i = 0; i < 8; ++i) z[i] = 0.0f;
  return z;
}

// --- CDNA5 data movers ------------------------------------------------------

// Async memory->LDS copy, 16B per lane. ASYNCcnt-tracked (ISA 10. / 15.18.3).
static __device__ __forceinline__ void async_copy_b128(unsigned lds_byte_addr,
                                                       const void* gaddr) {
  asm volatile("global_load_async_to_lds_b128 %0, %1, off"
               :: "v"(lds_byte_addr), "v"((unsigned long long)(size_t)gaddr)
               : "memory");
}

static __device__ __forceinline__ void wait_asynccnt0() {
#if __has_builtin(__builtin_amdgcn_s_wait_asynccnt)
  __builtin_amdgcn_s_wait_asynccnt(0);
#else
  asm volatile("s_wait_asynccnt 0x0" ::: "memory");
#endif
}

#if defined(HAVE_TDM)
// TDM: DMA a 64x64 bf16 tile (row stride 64 elements) into LDS, padding each
// 128B row with 16B so LDS row stride is 144B (= our [64][72] bf16 layout).
static __device__ __forceinline__ void tdm_load_tile64x64(unsigned lds_byte_addr,
                                                          const bf16* gsrc) {
  const unsigned long long ga = (unsigned long long)(size_t)gsrc;
  u32x4 g0;
  g0[0] = 1u;                                   // count=1, user descriptor
  g0[1] = lds_byte_addr;                        // lds_addr
  g0[2] = (unsigned)ga;                         // global_addr[31:0]
  g0[3] = (unsigned)((ga >> 32) & 0x1FFFFFFu) | (2u << 30);  // addr[56:32], type=2
  i32x8 g1;
  g1[0] = (int)0x07110000u;  // pad_amount=3(4dw) pad_interval=4(32dw) pad_en=1 dsz=1(2B)
  g1[1] = 64 << 16;          // tensor_dim0 = 64     (bits 79:48)
  g1[2] = 64 << 16;          // tensor_dim1 = 64     (bits 111:80)
  g1[3] = 64 << 16;          // tile_dim0   = 64     (bits 127:112)
  g1[4] = 64;                // tile_dim1   = 64     (bits 143:128)
  g1[5] = 64;                // tensor_dim0_stride = 64 (bits 207:160)
  g1[6] = 0;
  g1[7] = 0;
  const i32x4 z4 = {0, 0, 0, 0};
  const i32x8 z8 = {0, 0, 0, 0, 0, 0, 0, 0};
  __builtin_amdgcn_tensor_load_to_lds(g0, g1, z4, z4, z8, 0);
}

static __device__ __forceinline__ void wait_tensorcnt0() {
#if __has_builtin(__builtin_amdgcn_s_wait_tensorcnt)
  __builtin_amdgcn_s_wait_tensorcnt(0);
#else
  asm volatile("s_wait_tensorcnt 0x0" ::: "memory");
#endif
}
#endif

// --- WMMA fragment helpers --------------------------------------------------

// A fragment 16x32 bf16 from row-major source (stride in elements).
// Lanes 0-15 row M=lane, K={0..7,16..23}; lanes 16-31 K={8..15,24..31}.
static __device__ __forceinline__ v16bf load_a_frag(const bf16* a, int stride, int lane) {
  const int m = lane & 15, half = lane >> 4;
  const unsigned* rowu = (const unsigned*)(a + m * stride);
  v8u t;
#pragma unroll
  for (int i = 0; i < 4; ++i) t[i] = rowu[i + 4 * half];
#pragma unroll
  for (int i = 0; i < 4; ++i) t[4 + i] = rowu[8 + i + 4 * half];
  return __builtin_bit_cast(v16bf, t);
}

// B fragment 32x16 (KxN) bf16 from an N-major staging buffer bt[n][k].
static __device__ __forceinline__ v16bf load_b_frag(const bf16* bt, int stride, int lane) {
  const int n = lane & 15, half = lane >> 4;
  const unsigned* rowu = (const unsigned*)(bt + n * stride);
  v8u t;
#pragma unroll
  for (int j = 0; j < 8; ++j) t[j] = rowu[(j & 3) + 4 * half + 8 * (j >> 2)];
  return __builtin_bit_cast(v16bf, t);
}

static __device__ __forceinline__ v8f wmma_bf16(v16bf a, v16bf b, v8f c) {
  return __builtin_amdgcn_wmma_f32_16x16x32_bf16(false, a, false, b, (short)0, c,
                                                 false, false);
}

// ---------------------------------------------------------------------------
// f32 -> bf16 conversion (optionally scaled), 4 elements per thread iteration
// ---------------------------------------------------------------------------
__global__ void convert_bf16_kernel(const float* __restrict__ src, bf16* __restrict__ dst,
                                    int n4, float scale) {
  int i = blockIdx.x * blockDim.x + threadIdx.x;
  const int step = gridDim.x * blockDim.x;
  for (; i < n4; i += step) {
    const float4 f = ((const float4*)src)[i];
    union { uint2 q; bf16 e[4]; } o;
    o.e[0] = (bf16)(f.x * scale);
    o.e[1] = (bf16)(f.y * scale);
    o.e[2] = (bf16)(f.z * scale);
    o.e[3] = (bf16)(f.w * scale);
    ((uint2*)dst)[i] = o.q;
  }
}

// ---------------------------------------------------------------------------
// bf16 GEMM, C = A[M,K] @ B[K,N]; 128x128 block tile, 8 waves of 64x32.
// Double-buffered: A tile via async memory->LDS DMA, B tile via registers with
// packed-b32 transposed stores; next tile's traffic overlaps current WMMAs.
// MODE 1: store bf16 into [B,H,T,D] head layout. MODE 2: store f32 row-major.
// ---------------------------------------------------------------------------
template <int MODE>
__global__ __launch_bounds__(256) void gemm_bf16_kernel(const bf16* __restrict__ A,
                                                        const bf16* __restrict__ Bm,
                                                        void* __restrict__ Cout,
                                                        int M, int N, int K) {
  (void)M;
  __shared__ bf16 As[2][128][40];  // A tiles, row-major [m][k], padded stride
  __shared__ bf16 Bt[2][128][40];  // B tiles transposed [n][k], padded stride

  const int tid = threadIdx.x, lane = tid & 31, wave = tid >> 5;
  const int mw = wave & 1, nw = wave >> 1;
  const int m0 = blockIdx.y * 128, n0 = blockIdx.x * 128;
  const int ln = lane & 15, half = lane >> 4;

  v8f acc[4][2];
#pragma unroll
  for (int mt = 0; mt < 4; ++mt)
#pragma unroll
    for (int nt = 0; nt < 2; ++nt) acc[mt][nt] = vzero8();

  const int ar = tid >> 1, akh = (tid & 1) * 16;         // A: 16 bf16/thread
  const int bkp = (tid & 15) * 2, bnh = (tid >> 4) * 8;  // B: 2k x 8n per thread
  const unsigned ldsA0 = (unsigned)(size_t)&As[0][ar][akh];
  const unsigned ldsA1 = (unsigned)(size_t)&As[1][ar][akh];

  uint4 br0, br1;  // in-flight B tile (registers)

  auto issueA = [&](int k0, int buf) {
    const bf16* ga = A + (size_t)(m0 + ar) * K + k0 + akh;
    const unsigned l = buf ? ldsA1 : ldsA0;
    async_copy_b128(l, ga);
    async_copy_b128(l + 16, ga + 8);
  };
  auto loadB = [&](int k0) {
    br0 = *(const uint4*)(Bm + (size_t)(k0 + bkp) * N + n0 + bnh);
    br1 = *(const uint4*)(Bm + (size_t)(k0 + bkp + 1) * N + n0 + bnh);
  };
  auto storeB = [&](int buf) {
    union { uint4 q; bf16 e[8]; } u0, u1;
    u0.q = br0;
    u1.q = br1;
#pragma unroll
    for (int i = 0; i < 8; ++i) {
      union { bf16 e[2]; unsigned u; } p;
      p.e[0] = u0.e[i];
      p.e[1] = u1.e[i];
      *(unsigned*)&Bt[buf][bnh + i][bkp] = p.u;
    }
  };

  const int nsteps = K / 32;
  issueA(0, 0);
  loadB(0);

  for (int t = 0; t < nsteps; ++t) {
    const int cur = t & 1;
    wait_asynccnt0();   // As[cur] DMA complete (this wave's issues)
    storeB(cur);        // commit in-flight B regs to Bt[cur]
    __syncthreads();    // tiles[cur] visible; tiles[1-cur] free (read at t-1)
    if (t + 1 < nsteps) {
      issueA((t + 1) * 32, 1 - cur);  // DMA next A under the WMMAs
      loadB((t + 1) * 32);            // next B into registers under the WMMAs
      if (t + 2 < nsteps) {
        __builtin_prefetch(A + (size_t)(m0 + ar) * K + (t + 2) * 32 + akh, 0, 0);
        __builtin_prefetch(Bm + (size_t)((t + 2) * 32 + bkp) * N + n0 + bnh, 0, 0);
      }
    }

    v16bf af[4], bfr[2];
#pragma unroll
    for (int mt = 0; mt < 4; ++mt)
      af[mt] = load_a_frag(&As[cur][mw * 64 + mt * 16][0], 40, lane);
#pragma unroll
    for (int nt = 0; nt < 2; ++nt)
      bfr[nt] = load_b_frag(&Bt[cur][nw * 32 + nt * 16][0], 40, lane);
#pragma unroll
    for (int mt = 0; mt < 4; ++mt)
#pragma unroll
      for (int nt = 0; nt < 2; ++nt)
        acc[mt][nt] = wmma_bf16(af[mt], bfr[nt], acc[mt][nt]);
  }

  // Epilogue. C/D layout: lanes 0-15 -> M=r, lanes 16-31 -> M=r+8; N = ln.
#pragma unroll
  for (int mt = 0; mt < 4; ++mt)
#pragma unroll
    for (int nt = 0; nt < 2; ++nt)
#pragma unroll
      for (int r = 0; r < 8; ++r) {
        const int row = m0 + mw * 64 + mt * 16 + r + 8 * half;
        const int col = n0 + nw * 32 + nt * 16 + ln;
        const float v = acc[mt][nt][r];
        if (MODE == 1) {
          bf16* ob = (bf16*)Cout;
          const int b_ = row >> 11, t_ = row & (SEQ_T - 1);
          const int h_ = col >> 6, d_ = col & (HEAD_DIM - 1);
          ob[((size_t)(b_ * N_HEADS + h_) * SEQ_T + t_) * HEAD_DIM + d_] = (bf16)v;
        } else {
          float* of = (float*)Cout;
          of[(size_t)row * N + col] = v;
        }
      }
}

// ---------------------------------------------------------------------------
// Sliding-window flash attention. One WG (4 waves) per (b, h, 64-query block).
// Double-buffered K (TDM) and V (register-staged transpose); next key block's
// DMA and loads run under the current block's WMMAs + softmax.
// ---------------------------------------------------------------------------
__global__ __launch_bounds__(128) void attn_kernel(const bf16* __restrict__ Qh,
                                                   const bf16* __restrict__ Kh,
                                                   const bf16* __restrict__ Vh,
                                                   bf16* __restrict__ O) {
  __shared__ bf16 Qs[64][72];         // [q][d]
  __shared__ bf16 Ks[2][64][72];      // [key][d]
  __shared__ bf16 Vt[2][64][72];      // [d][key]
  __shared__ bf16 Ps[4][16][72];      // per-wave P tile [q][key]

  const int tid = threadIdx.x, lane = tid & 31, wave = tid >> 5;
  const int ln = lane & 15, half = lane >> 4;
  const int qb = blockIdx.x, h = blockIdx.y, b = blockIdx.z;
  const int q0 = qb * 64;
  const size_t headbase = (size_t)(b * N_HEADS + h) * SEQ_T * HEAD_DIM;
  const float NEG_INF = -__builtin_inff();

  const int kp = (tid & 31) * 2, vdh = (tid >> 5) * 16;  // V staging: 2 keys x 16 d
  uint4 vr[4];  // in-flight V rows (registers)

  auto loadV = [&](int kb) {
    const uint4* g0 = (const uint4*)(Vh + headbase + (size_t)(kb * 64 + kp) * HEAD_DIM + vdh);
    const uint4* g1 = (const uint4*)(Vh + headbase + (size_t)(kb * 64 + kp + 1) * HEAD_DIM + vdh);
    vr[0] = g0[0]; vr[1] = g0[1]; vr[2] = g1[0]; vr[3] = g1[1];
  };
  auto storeV = [&](int buf) {  // transpose, adjacent-key pairs packed into b32
    union { uint4 q[2]; bf16 e[16]; } r0, r1;
    r0.q[0] = vr[0]; r0.q[1] = vr[1];
    r1.q[0] = vr[2]; r1.q[1] = vr[3];
#pragma unroll
    for (int i = 0; i < 16; ++i) {
      union { bf16 e[2]; unsigned u; } p;
      p.e[0] = r0.e[i];
      p.e[1] = r1.e[i];
      *(unsigned*)&Vt[buf][vdh + i][kp] = p.u;
    }
  };

  const int kb_lo = (qb >= 8) ? (qb - 8) : 0;  // keys [q0-512, q0+63]

  // Prologue: stage Q and first K tile; first V into registers.
#if defined(HAVE_TDM)
  if (wave == 0) {
    tdm_load_tile64x64((unsigned)(size_t)&Qs[0][0],
                       Qh + headbase + (size_t)q0 * HEAD_DIM);
    tdm_load_tile64x64((unsigned)(size_t)&Ks[0][0][0],
                       Kh + headbase + (size_t)(kb_lo * 64) * HEAD_DIM);
  }
#else
  {
    const int r = tid >> 1, dh = (tid & 1) * 32;
    const uint4* gq = (const uint4*)(Qh + headbase + (size_t)(q0 + r) * HEAD_DIM + dh);
    const uint4* gk = (const uint4*)(Kh + headbase + (size_t)(kb_lo * 64 + r) * HEAD_DIM + dh);
#pragma unroll
    for (int i = 0; i < 4; ++i) *(uint4*)&Qs[r][dh + 8 * i] = gq[i];
#pragma unroll
    for (int i = 0; i < 4; ++i) *(uint4*)&Ks[0][r][dh + 8 * i] = gk[i];
  }
#endif
  loadV(kb_lo);
#if defined(HAVE_TDM)
  if (wave == 0) wait_tensorcnt0();
#endif
  __syncthreads();

  v16bf qf[2];
#pragma unroll
  for (int ks = 0; ks < 2; ++ks)
    qf[ks] = load_a_frag(&Qs[wave * 16][ks * 32], 72, lane);

  v8f oacc[4];
#pragma unroll
  for (int dt = 0; dt < 4; ++dt) oacc[dt] = vzero8();
  float mrow[8], lrow[8];
#pragma unroll
  for (int r = 0; r < 8; ++r) { mrow[r] = NEG_INF; lrow[r] = 0.0f; }

  for (int kb = kb_lo; kb <= qb; ++kb) {
    const int cur = (kb - kb_lo) & 1;
    storeV(cur);  // commit in-flight V regs (buffer free since barrier kb-1)
#if defined(HAVE_TDM)
    if (wave == 0) wait_tensorcnt0();  // K[cur] landed (no-op on first iter)
#endif
    __syncthreads();
    if (kb < qb) {  // stage next block under this block's compute
#if defined(HAVE_TDM)
      if (wave == 0)
        tdm_load_tile64x64((unsigned)(size_t)&Ks[1 - cur][0][0],
                           Kh + headbase + (size_t)((kb + 1) * 64) * HEAD_DIM);
#else
      {
        const int r = tid >> 1, dh = (tid & 1) * 32;
        const uint4* gk =
            (const uint4*)(Kh + headbase + (size_t)((kb + 1) * 64 + r) * HEAD_DIM + dh);
#pragma unroll
        for (int i = 0; i < 4; ++i) *(uint4*)&Ks[1 - cur][r][dh + 8 * i] = gk[i];
      }
#endif
      loadV(kb + 1);
      if (kb + 2 <= qb)
        __builtin_prefetch(Vh + headbase + (size_t)((kb + 2) * 64 + kp) * HEAD_DIM + vdh,
                           0, 0);
    }

    // S = Q * K^T   (wave's 16 rows x 64 keys)
    v8f sacc[4];
#pragma unroll
    for (int nt = 0; nt < 4; ++nt) sacc[nt] = vzero8();
#pragma unroll
    for (int nt = 0; nt < 4; ++nt)
#pragma unroll
      for (int ks = 0; ks < 2; ++ks)
        sacc[nt] =
            wmma_bf16(qf[ks], load_b_frag(&Ks[cur][nt * 16][ks * 32], 72, lane), sacc[nt]);

    // causal + window mask: valid iff (j <= i) && (j > i - WINDOW)
#pragma unroll
    for (int nt = 0; nt < 4; ++nt) {
      const int j = kb * 64 + nt * 16 + ln;
#pragma unroll
      for (int r = 0; r < 8; ++r) {
        const int qi = q0 + wave * 16 + r + 8 * half;
        if (j > qi || j <= qi - WINDOW) sacc[nt][r] = NEG_INF;
      }
    }

    // online softmax (row stats via intra-half shuffles)
    float mnew[8], al[8], bs[8];
#pragma unroll
    for (int r = 0; r < 8; ++r) {
      float bm = fmaxf(fmaxf(sacc[0][r], sacc[1][r]), fmaxf(sacc[2][r], sacc[3][r]));
#pragma unroll
      for (int off = 8; off >= 1; off >>= 1) bm = fmaxf(bm, __shfl_xor(bm, off));
      mnew[r] = fmaxf(mrow[r], bm);
      al[r] = (mrow[r] == mnew[r]) ? 1.0f : exp2f(mrow[r] - mnew[r]);
      mrow[r] = mnew[r];
      bs[r] = 0.0f;
    }

#pragma unroll
    for (int nt = 0; nt < 4; ++nt)
#pragma unroll
      for (int r = 0; r < 8; ++r) {
        const float e = sacc[nt][r] - mnew[r];
        const float p = (e <= 0.0f) ? exp2f(e) : 0.0f;  // NaN/inf-inf safe
        bs[r] += p;
        Ps[wave][r + 8 * half][nt * 16 + ln] = (bf16)p;
      }

#pragma unroll
    for (int r = 0; r < 8; ++r) {
#pragma unroll
      for (int off = 8; off >= 1; off >>= 1) bs[r] += __shfl_xor(bs[r], off);
      lrow[r] = lrow[r] * al[r] + bs[r];
    }
#pragma unroll
    for (int dt = 0; dt < 4; ++dt)
#pragma unroll
      for (int r = 0; r < 8; ++r) oacc[dt][r] *= al[r];

    // O += P * V  (same-wave LDS store->load is in-order)
    v16bf pa[2];
#pragma unroll
    for (int ks = 0; ks < 2; ++ks)
      pa[ks] = load_a_frag(&Ps[wave][0][ks * 32], 72, lane);
#pragma unroll
    for (int dt = 0; dt < 4; ++dt)
#pragma unroll
      for (int ks = 0; ks < 2; ++ks)
        oacc[dt] =
            wmma_bf16(pa[ks], load_b_frag(&Vt[cur][dt * 16][ks * 32], 72, lane), oacc[dt]);
  }

  // write attn_out [M_TOTAL, N_EMBD] (bf16), row = b*T + t, col = h*64 + d
#pragma unroll
  for (int dt = 0; dt < 4; ++dt)
#pragma unroll
    for (int r = 0; r < 8; ++r) {
      const int t_glob = q0 + wave * 16 + r + 8 * half;
      const int col = h * HEAD_DIM + dt * 16 + ln;
      const float v = oacc[dt][r] / lrow[r];
      O[(size_t)(b * SEQ_T + t_glob) * N_EMBD + col] = (bf16)v;
    }
}

// ---------------------------------------------------------------------------
extern "C" void kernel_launch(void* const* d_in, const int* in_sizes, int n_in,
                              void* d_out, int out_size, void* d_ws, size_t ws_size,
                              hipStream_t stream) {
  (void)in_sizes; (void)n_in; (void)out_size; (void)ws_size;
  const float* x  = (const float*)d_in[0];
  const float* Wq = (const float*)d_in[1];
  const float* Wk = (const float*)d_in[2];
  const float* Wv = (const float*)d_in[3];
  const float* Wo = (const float*)d_in[4];

  char* ws = (char*)d_ws;  // ~48 MB used
  bf16* xb   = (bf16*)(ws);
  bf16* wqb  = (bf16*)(ws + (size_t)8  * 1024 * 1024);
  bf16* wkb  = (bf16*)(ws + (size_t)10 * 1024 * 1024);
  bf16* wvb  = (bf16*)(ws + (size_t)12 * 1024 * 1024);
  bf16* wob  = (bf16*)(ws + (size_t)14 * 1024 * 1024);
  bf16* Qh   = (bf16*)(ws + (size_t)16 * 1024 * 1024);
  bf16* Kh   = (bf16*)(ws + (size_t)24 * 1024 * 1024);
  bf16* Vh   = (bf16*)(ws + (size_t)32 * 1024 * 1024);
  bf16* attn = (bf16*)(ws + (size_t)40 * 1024 * 1024);

  // fold softmax scale (1/sqrt(64)) and log2(e) into Wq -> softmax is exp2-domain
  const float QSCALE = 0.125f * 1.44269504088896340736f;

  convert_bf16_kernel<<<1024, 256, 0, stream>>>(x,  xb,  (M_TOTAL * N_EMBD) / 4, 1.0f);
  convert_bf16_kernel<<<512, 256, 0, stream>>>(Wq, wqb, (N_EMBD * N_EMBD) / 4, QSCALE);
  convert_bf16_kernel<<<512, 256, 0, stream>>>(Wk, wkb, (N_EMBD * N_EMBD) / 4, 1.0f);
  convert_bf16_kernel<<<512, 256, 0, stream>>>(Wv, wvb, (N_EMBD * N_EMBD) / 4, 1.0f);
  convert_bf16_kernel<<<512, 256, 0, stream>>>(Wo, wob, (N_EMBD * N_EMBD) / 4, 1.0f);

  dim3 ggrid(N_EMBD / 128, M_TOTAL / 128);
  gemm_bf16_kernel<1><<<ggrid, 256, 0, stream>>>(xb, wqb, (void*)Qh, M_TOTAL, N_EMBD, N_EMBD);
  gemm_bf16_kernel<1><<<ggrid, 256, 0, stream>>>(xb, wkb, (void*)Kh, M_TOTAL, N_EMBD, N_EMBD);
  gemm_bf16_kernel<1><<<ggrid, 256, 0, stream>>>(xb, wvb, (void*)Vh, M_TOTAL, N_EMBD, N_EMBD);

  dim3 agrid(SEQ_T / 64, N_HEADS, BATCH);
  attn_kernel<<<agrid, 128, 0, stream>>>(Qh, Kh, Vh, attn);

  gemm_bf16_kernel<2><<<ggrid, 256, 0, stream>>>(attn, wob, d_out, M_TOTAL, N_EMBD, N_EMBD);
}
